// feat_embedding_layer_9534827397769
// MI455X (gfx1250) — compile-verified, matrix operand order checked by reading, not run on hardware
//
#include <hip/hip_runtime.h>

typedef __attribute__((ext_vector_type(16))) _Float16 v16h;
typedef __attribute__((ext_vector_type(8)))  _Float16 v8h;
typedef __attribute__((ext_vector_type(8)))  float    v8f;

#define WAVES_PER_BLOCK 8

static __device__ inline v8f wmma_f16(v16h a, v16h b, v8f c) {
  // (neg_a, A, neg_b, B, c_mod, C, reuse_a, reuse_b)
  return __builtin_amdgcn_wmma_f32_16x16x32_f16(false, a, false, b, (short)0, c,
                                                false, false);
}

// Build one 16-half fragment for the f16 16x16x32 WMMA.
// Per ISA 7.12.2 (16-bit A / B layout): halves [0,8) hold K = KOFS+kb+j,
// halves [8,16) hold K = KOFS+16+kb+j, kb = 8*(lane>=16). Zero beyond KMAX.
template <int KOFS, int KMAX>
static __device__ inline v16h gather_frag(const float* __restrict__ base, int kb) {
  v16h v;
#pragma unroll
  for (int j = 0; j < 8; ++j) {
    const int k0 = KOFS + kb + j;
    const int k1 = KOFS + 16 + kb + j;
    const float f0 = (k0 < KMAX) ? base[k0] : 0.0f;
    const float f1 = (k1 < KMAX) ? base[k1] : 0.0f;
    v[j]     = (_Float16)f0;
    v[j + 8] = (_Float16)f1;
  }
  return v;
}

// Read an A-fragment (16 rows x 32 K, f16) from a row-major 32-col LDS tile.
// Two aligned 16B runs per lane -> ds_load_b128 pairs.
static __device__ inline v16h lds_afrag(const _Float16* __restrict__ h, int row, int kb) {
  const v8h lo = *(const v8h*)(h + row * 32 + kb);
  const v8h hv = *(const v8h*)(h + row * 32 + 16 + kb);
  v16h a;
#pragma unroll
  for (int j = 0; j < 8; ++j) { a[j] = lo[j]; a[j + 8] = hv[j]; }
  return a;
}

// Write a C-fragment (f32 accum) to the LDS tile as f16.
// C layout: VGPR r holds element (M = rowbase + r, N = col).
static __device__ inline void lds_store_c(_Float16* __restrict__ h, v8f c,
                                          int rowbase, int col) {
#pragma unroll
  for (int r = 0; r < 8; ++r) h[(rowbase + r) * 32 + col] = (_Float16)c[r];
}

__global__ __launch_bounds__(256) void pn_mlp_max_kernel(
    const float* __restrict__ X,
    const float* __restrict__ W1, const float* __restrict__ B1,
    const float* __restrict__ W2, const float* __restrict__ B2,
    const float* __restrict__ W3, const float* __restrict__ B3,
    float* __restrict__ out, int G, int totalWaves) {
  __shared__ __align__(16) _Float16 lds_h[WAVES_PER_BLOCK][32 * 32];

  const int lane = threadIdx.x & 31;
  const int wib  = threadIdx.x >> 5;
  _Float16* hbuf = &lds_h[wib][0];

  const int ln = lane & 15;        // column / row selector within a 16-tile
  const int hi = lane >> 4;        // 0: low half, 1: high half
  const int kb = hi * 8;

  // ---- Weight B-fragments, built once per wave (B[k][n] = W[n][k]) ----
  v16h bw1[2][2];  // [kstep][ntile]
  v16h bw2[2], bw3[2];
#pragma unroll
  for (int nt = 0; nt < 2; ++nt) {
    const int n = ln + nt * 16;
    bw1[0][nt] = gather_frag<0, 35>(W1 + n * 35, kb);
    bw1[1][nt] = gather_frag<32, 35>(W1 + n * 35, kb);
    bw2[nt]    = gather_frag<0, 32>(W2 + n * 32, kb);
    bw3[nt]    = gather_frag<0, 32>(W3 + n * 32, kb);
  }
  float bias1[2], bias2[2], bias3[2];
#pragma unroll
  for (int nt = 0; nt < 2; ++nt) {
    bias1[nt] = B1[ln + nt * 16];
    bias2[nt] = B2[ln + nt * 16];
    bias3[nt] = B3[ln + nt * 16];
  }

  const int gw = blockIdx.x * WAVES_PER_BLOCK + wib;
  for (int g = gw; g < G; g += totalWaves) {
    const float* xg = X + (size_t)g * (32 * 35);
    __builtin_prefetch(xg + (size_t)totalWaves * (32 * 35), 0, 0);

    // ---------------- Layer 1: (32x35) @ (35x32), F padded to 64 ----------
    v8f c1[2][2];
#pragma unroll
    for (int rt = 0; rt < 2; ++rt)
#pragma unroll
      for (int nt = 0; nt < 2; ++nt)
#pragma unroll
        for (int r = 0; r < 8; ++r) c1[rt][nt][r] = bias1[nt];

#pragma unroll
    for (int rt = 0; rt < 2; ++rt) {
      const float* rowp = xg + (size_t)(rt * 16 + ln) * 35;
      const v16h a0 = gather_frag<0, 35>(rowp, kb);
      const v16h a1 = gather_frag<32, 35>(rowp, kb);
#pragma unroll
      for (int nt = 0; nt < 2; ++nt) {
        c1[rt][nt] = wmma_f16(a0, bw1[0][nt], c1[rt][nt]);
        c1[rt][nt] = wmma_f16(a1, bw1[1][nt], c1[rt][nt]);
      }
    }
#pragma unroll
    for (int rt = 0; rt < 2; ++rt)
#pragma unroll
      for (int nt = 0; nt < 2; ++nt)
        lds_store_c(hbuf, c1[rt][nt], rt * 16 + hi * 8, ln + nt * 16);
    asm volatile("s_wait_dscnt 0" ::: "memory");

    // ---------------- Layer 2: (32x32) @ (32x32) --------------------------
    v8f c2[2][2];
#pragma unroll
    for (int rt = 0; rt < 2; ++rt)
#pragma unroll
      for (int nt = 0; nt < 2; ++nt)
#pragma unroll
        for (int r = 0; r < 8; ++r) c2[rt][nt][r] = bias2[nt];

#pragma unroll
    for (int rt = 0; rt < 2; ++rt) {
      const v16h a = lds_afrag(hbuf, rt * 16 + ln, kb);
#pragma unroll
      for (int nt = 0; nt < 2; ++nt)
        c2[rt][nt] = wmma_f16(a, bw2[nt], c2[rt][nt]);
    }
#pragma unroll
    for (int rt = 0; rt < 2; ++rt)
#pragma unroll
      for (int nt = 0; nt < 2; ++nt)
        lds_store_c(hbuf, c2[rt][nt], rt * 16 + hi * 8, ln + nt * 16);
    asm volatile("s_wait_dscnt 0" ::: "memory");

    // ---------------- Layer 3 + max over the 32 neighbor rows -------------
    v8f c3[2][2];
#pragma unroll
    for (int rt = 0; rt < 2; ++rt)
#pragma unroll
      for (int nt = 0; nt < 2; ++nt)
#pragma unroll
        for (int r = 0; r < 8; ++r) c3[rt][nt][r] = bias3[nt];

#pragma unroll
    for (int rt = 0; rt < 2; ++rt) {
      const v16h a = lds_afrag(hbuf, rt * 16 + ln, kb);
#pragma unroll
      for (int nt = 0; nt < 2; ++nt)
        c3[rt][nt] = wmma_f16(a, bw3[nt], c3[rt][nt]);
    }

    float m[2];
#pragma unroll
    for (int nt = 0; nt < 2; ++nt) {
      float v = fmaxf(c3[0][nt][0], c3[1][nt][0]);
#pragma unroll
      for (int r = 1; r < 8; ++r)
        v = fmaxf(v, fmaxf(c3[0][nt][r], c3[1][nt][r]));
      // combine M-halves held in lane <-> lane^16
      v = fmaxf(v, __shfl_xor(v, 16, 32));
      m[nt] = v;
    }
    // lane 0..15 stores N=lane (ntile 0), lane 16..31 stores N=lane (ntile 1)
    out[(size_t)g * 32 + lane] = hi ? m[1] : m[0];
  }
}

extern "C" void kernel_launch(void* const* d_in, const int* in_sizes, int n_in,
                              void* d_out, int out_size, void* d_ws, size_t ws_size,
                              hipStream_t stream) {
  (void)n_in; (void)d_ws; (void)ws_size; (void)out_size;
  const float* X  = (const float*)d_in[0];
  const float* W1 = (const float*)d_in[1];
  const float* b1 = (const float*)d_in[2];
  const float* W2 = (const float*)d_in[3];
  const float* b2 = (const float*)d_in[4];
  const float* W3 = (const float*)d_in[5];
  const float* b3 = (const float*)d_in[6];
  float* out = (float*)d_out;

  const int G = in_sizes[0] / (32 * 35);  // B*N groups of 32 points
  // ~8 groups per wave to amortize the per-wave weight-fragment build.
  int blocks = (G + WAVES_PER_BLOCK * 8 - 1) / (WAVES_PER_BLOCK * 8);
  if (blocks < 1) blocks = 1;
  if (blocks > 4096) blocks = 4096;
  const int totalWaves = blocks * WAVES_PER_BLOCK;

  pn_mlp_max_kernel<<<blocks, 256, 0, stream>>>(X, W1, b1, W2, b2, W3, b3,
                                                out, G, totalWaves);
}
